// ContrastiveLoss_27728308863729
// MI455X (gfx1250) — compile-verified, matrix operand order checked by reading, not run on hardware
//
#include <hip/hip_runtime.h>

#define NB   16384
#define NDIM 256
#define EPSN 1e-8f

typedef __attribute__((ext_vector_type(16))) __bf16       v16bf;
typedef __attribute__((ext_vector_type(8)))  float        v8f;
typedef __attribute__((ext_vector_type(4)))  float        v4f;
typedef __attribute__((ext_vector_type(4)))  unsigned int v4u;
typedef __attribute__((ext_vector_type(4)))  int          v4i;

union FragU { v16bf v; v4u u4[2]; };
union PackU { __bf16 h[8]; v4u u; };

// Use the gfx1250 async Global->LDS DMA path when the toolchain exposes it.
#if defined(__has_builtin)
#if __has_builtin(__builtin_amdgcn_global_load_async_to_lds_b128)
#define USE_ASYNC_LDS 1
#endif
#endif

#ifdef USE_ASYNC_LDS
#define WAIT_ASYNC() asm volatile("s_wait_asynccnt 0" ::: "memory")
typedef __attribute__((address_space(1))) v4i* g_v4i_p;   // global int4*
typedef __attribute__((address_space(3))) v4i* l_v4i_p;   // LDS int4*
#else
#define WAIT_ASYNC()
#endif

// ---------------------------------------------------------------------------
// Phase 1: row normalization (f32) + exact diagonal dot, emit bf16 operands.
// One wave (32 lanes) per row; lane handles 8 of 256 elements.
// ---------------------------------------------------------------------------
__global__ __launch_bounds__(256) void cl_normalize(
    const float* __restrict__ T, const float* __restrict__ S,
    __bf16* __restrict__ Tn, __bf16* __restrict__ Sn,
    float* __restrict__ posdot)
{
  const int wave = threadIdx.x >> 5;
  const int lane = threadIdx.x & 31;
  const int row  = blockIdx.x * 8 + wave;

  const v4f* tp = (const v4f*)(T + (size_t)row * NDIM) + lane * 2;
  const v4f* sp = (const v4f*)(S + (size_t)row * NDIM) + lane * 2;
  v4f t0 = tp[0], t1 = tp[1], s0 = sp[0], s1 = sp[1];

  float tv[8], sv[8];
  #pragma unroll
  for (int i = 0; i < 4; ++i) { tv[i] = t0[i]; tv[4+i] = t1[i]; sv[i] = s0[i]; sv[4+i] = s1[i]; }

  float tt = 0.f, ss = 0.f, ts = 0.f;
  #pragma unroll
  for (int i = 0; i < 8; ++i) {
    tt = fmaf(tv[i], tv[i], tt);
    ss = fmaf(sv[i], sv[i], ss);
    ts = fmaf(tv[i], sv[i], ts);
  }
  #pragma unroll
  for (int m = 16; m >= 1; m >>= 1) {
    tt += __shfl_xor(tt, m);
    ss += __shfl_xor(ss, m);
    ts += __shfl_xor(ts, m);
  }
  const float invT = 1.0f / fmaxf(sqrtf(tt), EPSN);
  const float invS = 1.0f / fmaxf(sqrtf(ss), EPSN);
  if (lane == 0) posdot[row] = ts * invT * invS;   // exact f32 diagonal term

  PackU pt, ps;
  #pragma unroll
  for (int i = 0; i < 8; ++i) {
    pt.h[i] = (__bf16)(tv[i] * invT);
    ps.h[i] = (__bf16)(sv[i] * invS);
  }
  *(v4u*)(Tn + (size_t)row * NDIM + lane * 8) = pt.u;
  *(v4u*)(Sn + (size_t)row * NDIM + lane * 8) = ps.u;
}

// ---------------------------------------------------------------------------
// Phase 2: fused bf16-WMMA GEMM  sim = Tn * Sn^T  with exp + label-mask
// row-reduction.
//
// 8 waves / block; each wave owns a 16-row strip with its 8 A fragments
// (K = 256) resident in 64 VGPRs. Column strips of 32 rows of Sn are staged
// into LDS once per workgroup (shared by all 8 waves) via the gfx1250 async
// Global->LDS DMA, double-buffered so the next strip overlaps the current
// strip's 16 WMMAs. LDS rows are padded to 528 B (= 16 mod 256) so the 16
// N-lanes of each half-wave hit disjoint 4-bank groups on ds_load_b128.
//
// 16-bit fragment layout per ISA: lane L(0-15) = row L holds K {0..7} in
// V0-3 and K {16..23} in V4-7; lane L+16 holds K {8..15},{24..31}. Because
// sim = Tn * Sn^T with both row-major, A and B fragments use identical
// per-lane addressing.
// ---------------------------------------------------------------------------
__global__ __launch_bounds__(256) void cl_simexp(
    const __bf16* __restrict__ Tn, const __bf16* __restrict__ Sn,
    const int* __restrict__ labels, float* __restrict__ neg)
{
  constexpr int STAGE_COLS  = 32;                       // Sn rows per stage
  constexpr int LDS_ROW_B   = 264 * 2;                  // 528 bytes (256 bf16 + 8 pad)
  constexpr int STAGE_BYTES = STAGE_COLS * LDS_ROW_B;   // 16896
  constexpr int NSTAGES     = NB / STAGE_COLS;          // 512

  __shared__ __align__(16) char smem[2 * STAGE_BYTES];

  const int tid    = threadIdx.x;
  const int wave   = tid >> 5;
  const int lane   = tid & 31;
  const int laneLo = lane & 15;
  const int laneHi = lane >> 4;
  const int rA     = blockIdx.x * 128 + wave * 16;      // first row of this wave

  // Preload all A fragments for K = 0..255 (8 frags x 8 VGPRs = 64 VGPRs).
  const __bf16* pA = Tn + (size_t)(rA + laneLo) * NDIM + 8 * laneHi;
  v16bf afrag[8];
  #pragma unroll
  for (int kk = 0; kk < 8; ++kk) {
    FragU u;
    u.u4[0] = *(const v4u*)(pA + kk * 32);
    u.u4[1] = *(const v4u*)(pA + kk * 32 + 16);
    afrag[kk] = u.v;
  }

  // Row labels for the 8 M-values this lane owns in the C fragment:
  // C layout: VGPR r, lanes 0-15 -> M=r ; lanes 16-31 -> M=8+r.
  int rl[8];
  #pragma unroll
  for (int r = 0; r < 8; ++r) rl[r] = labels[rA + 8 * laneHi + r];

  float acc[8];
  #pragma unroll
  for (int r = 0; r < 8; ++r) acc[r] = 0.f;

  // Per-thread staging addresses: thread covers rows (tid>>5)+8j of the
  // stage, 16-byte chunk (tid&31) within the 512-byte row, for j = 0..3.
  const int ccol   = tid >> 5;   // 0..7
  const int cchunk = tid & 31;   // 0..31
  const __bf16* gB0 = Sn + (size_t)ccol * NDIM + cchunk * 8;
  char*         lB0 = smem + ccol * LDS_ROW_B + cchunk * 16;

  auto issueStage = [&](int st, int buf) {
    const __bf16* g = gB0 + (size_t)st * STAGE_COLS * NDIM;
    char*         l = lB0 + buf * STAGE_BYTES;
    #pragma unroll
    for (int j = 0; j < 4; ++j) {
#ifdef USE_ASYNC_LDS
      __builtin_amdgcn_global_load_async_to_lds_b128(
          (g_v4i_p)(void*)(g + j * 8 * NDIM),
          (l_v4i_p)(void*)(l + j * 8 * LDS_ROW_B),
          0, 0);
#else
      *(v4u*)(l + j * 8 * LDS_ROW_B) = *(const v4u*)(g + j * 8 * NDIM);
#endif
    }
  };

  issueStage(0, 0);
  WAIT_ASYNC();
  __syncthreads();

  for (int st = 0; st < NSTAGES; ++st) {
    const int buf = st & 1;
    if (st + 1 < NSTAGES) issueStage(st + 1, buf ^ 1);   // overlap DMA w/ WMMA

    const char* ldsB = smem + buf * STAGE_BYTES;
    #pragma unroll
    for (int sub = 0; sub < 2; ++sub) {
      const int colBase = st * STAGE_COLS + sub * 16;
      const char* p = ldsB + (sub * 16 + laneLo) * LDS_ROW_B + laneHi * 16;

      v8f c = {};
      #pragma unroll
      for (int kk = 0; kk < 8; ++kk) {
        FragU u;
        u.u4[0] = *(const v4u*)(p + kk * 64);
        u.u4[1] = *(const v4u*)(p + kk * 64 + 32);
        c = __builtin_amdgcn_wmma_f32_16x16x32_bf16(
                /*neg_a=*/false, afrag[kk], /*neg_b=*/false, u.v,
                /*c_mod=*/(short)0, c, /*reuse_a=*/false, /*reuse_b=*/false);
      }

      const int cl = labels[colBase + laneLo];   // N = lane&15 for both halves
      #pragma unroll
      for (int r = 0; r < 8; ++r) {
        float e = __expf(c[r]);
        acc[r] += (rl[r] != cl) ? e : 0.0f;
      }
    }

    WAIT_ASYNC();      // own async stores to LDS complete before signaling
    __syncthreads();   // all waves done with buf before it is overwritten
  }

  // Reduce over the 16 N-lanes of each half-wave (masks < 16 never cross).
  #pragma unroll
  for (int r = 0; r < 8; ++r) {
    #pragma unroll
    for (int m = 8; m >= 1; m >>= 1) acc[r] += __shfl_xor(acc[r], m);
  }
  if (laneLo == 0) {
    #pragma unroll
    for (int r = 0; r < 8; ++r) neg[rA + 8 * laneHi + r] = acc[r];
  }
}

// ---------------------------------------------------------------------------
// Phase 3: loss = -(1/B) * sum_b ( pos_dot[b] - log(neg[b]) )
// ---------------------------------------------------------------------------
__global__ __launch_bounds__(1024) void cl_finalize(
    const float* __restrict__ posdot, const float* __restrict__ neg,
    float* __restrict__ out)
{
  __shared__ float red[32];
  float a = 0.f;
  for (int b = threadIdx.x; b < NB; b += 1024)
    a += posdot[b] - logf(neg[b]);
  #pragma unroll
  for (int m = 16; m >= 1; m >>= 1) a += __shfl_xor(a, m);
  if ((threadIdx.x & 31) == 0) red[threadIdx.x >> 5] = a;
  __syncthreads();
  if (threadIdx.x < 32) {
    float v = red[threadIdx.x];
    #pragma unroll
    for (int m = 16; m >= 1; m >>= 1) v += __shfl_xor(v, m);
    if (threadIdx.x == 0) out[0] = -v / (float)NB;
  }
}

extern "C" void kernel_launch(void* const* d_in, const int* in_sizes, int n_in,
                              void* d_out, int out_size, void* d_ws, size_t ws_size,
                              hipStream_t stream) {
  const float* T      = (const float*)d_in[0];
  const float* S      = (const float*)d_in[1];
  const int*   labels = (const int*)d_in[2];
  float*       out    = (float*)d_out;

  char* w = (char*)d_ws;
  const size_t nbf = (size_t)NB * NDIM * sizeof(__bf16);   // 8 MB each
  __bf16* Tn     = (__bf16*)w;
  __bf16* Sn     = (__bf16*)(w + nbf);
  float*  posdot = (float*)(w + 2 * nbf);
  float*  neg    = posdot + NB;

  cl_normalize<<<NB / 8,   256,  0, stream>>>(T, S, Tn, Sn, posdot);
  cl_simexp   <<<NB / 128, 256,  0, stream>>>(Tn, Sn, labels, neg);
  cl_finalize <<<1,        1024, 0, stream>>>(posdot, neg, out);
}
